// MultiHeadAttention_7834020348250
// MI455X (gfx1250) — matrix operand order repaired and checked by reading, not compiled, not run on hardware
//
#include <hip/hip_runtime.h>

// ---------------------------------------------------------------------------
// MultiHeadAttention forward for MI455X (gfx1250, wave32, WMMA bf16,
// async global->LDS DMA with ASYNCcnt split waits).
// B=4, H=16, S=2048, D=64, E=1024. fp32 in/out, bf16 WMMA w/ fp32 accumulate.
// ---------------------------------------------------------------------------

#define BATCH 4
#define SEQ   2048
#define EMBED 1024
#define NUMH  16
#define HEADD 64

typedef __bf16 bf16;
typedef __attribute__((ext_vector_type(16))) __bf16 v16bf;
typedef __attribute__((ext_vector_type(8)))  __bf16 bf16x8;
typedef __attribute__((ext_vector_type(8)))  float  v8f;

union frag16 { v16bf v; bf16x8 h[2]; };

// LDS byte offset of a __shared__ object: generic LDS addresses carry the
// LDS offset in addr[31:0] (flat-aperture rule), which is what the async
// instructions' VDST operand wants.
static __device__ __forceinline__ unsigned lds_off(const void* p) {
    return (unsigned)(unsigned long long)p;
}

// 2 x 16B async global->LDS (32 contiguous bytes); offset applies to both sides
#define ASYNC_COPY_32B(dst, src)                                        \
    asm volatile("global_load_async_to_lds_b128 %0, %1, off\n\t"        \
                 "global_load_async_to_lds_b128 %0, %1, off offset:16"  \
                 :: "v"(dst), "v"(src) : "memory")

// 8 x 16B async global->LDS (128 contiguous bytes)
#define ASYNC_COPY_128B(dst, src)                                        \
    asm volatile("global_load_async_to_lds_b128 %0, %1, off\n\t"         \
                 "global_load_async_to_lds_b128 %0, %1, off offset:16\n\t" \
                 "global_load_async_to_lds_b128 %0, %1, off offset:32\n\t" \
                 "global_load_async_to_lds_b128 %0, %1, off offset:48\n\t" \
                 "global_load_async_to_lds_b128 %0, %1, off offset:64\n\t" \
                 "global_load_async_to_lds_b128 %0, %1, off offset:80\n\t" \
                 "global_load_async_to_lds_b128 %0, %1, off offset:96\n\t" \
                 "global_load_async_to_lds_b128 %0, %1, off offset:112"  \
                 :: "v"(dst), "v"(src) : "memory")

#define WAIT_ASYNC(n) asm volatile("s_wait_asynccnt " #n ::: "memory")

static __device__ __forceinline__ v8f vzero() {
    v8f z;
#pragma unroll
    for (int i = 0; i < 8; ++i) z[i] = 0.0f;
    return z;
}

static __device__ __forceinline__ v8f wmma_bf16(const v16bf& a, const v16bf& b, const v8f& c) {
    return __builtin_amdgcn_wmma_f32_16x16x32_bf16(false, a, false, b, (short)0, c, false, false);
}

// max/sum reduction across the 16 lanes of a half-wave (lanes 0-15 and 16-31
// reduce independently; xor masks 1,2,4,8 never cross the 16-lane boundary)
static __device__ __forceinline__ float red_max16(float v) {
    v = fmaxf(v, __shfl_xor(v, 1, 32));
    v = fmaxf(v, __shfl_xor(v, 2, 32));
    v = fmaxf(v, __shfl_xor(v, 4, 32));
    v = fmaxf(v, __shfl_xor(v, 8, 32));
    return v;
}
static __device__ __forceinline__ float red_sum16(float v) {
    v += __shfl_xor(v, 1, 32);
    v += __shfl_xor(v, 2, 32);
    v += __shfl_xor(v, 4, 32);
    v += __shfl_xor(v, 8, 32);
    return v;
}

// ---------------------------------------------------------------------------
// fp32 -> bf16 cast
// ---------------------------------------------------------------------------
__global__ __launch_bounds__(256) void cast_f32_bf16_kernel(const float* __restrict__ x,
                                                            bf16* __restrict__ y, int n) {
    int i = blockIdx.x * 256 + threadIdx.x;
    if (i < n) y[i] = (bf16)x[i];
}

// ---------------------------------------------------------------------------
// Tiled 128x64 WMMA GEMM block: C[128x64] = alpha * A[128xK] * B[Kx64] (+bias)
// 8 waves, each wave produces 16 rows x 64 cols (4 C tiles of 16x16).
// A tile is DMA'd into double-buffered LDS via async loads (ASYNCcnt);
// B tile is loaded to VGPRs and stored transposed into double-buffered LDS.
// One barrier per K-step; DMA of step s+1 overlaps WMMAs of step s.
// ---------------------------------------------------------------------------
template <typename OutT, bool HAS_BIAS>
static __device__ __forceinline__ void gemm_block_128x64(
    const bf16* __restrict__ Ablk, int lda,
    const bf16* __restrict__ Bm,   int ldb,
    OutT* __restrict__ Cblk,       int ldc,
    int K, const float* __restrict__ bias, float alpha)
{
    __shared__ __attribute__((aligned(16))) bf16 As[2][128][32];
    __shared__ __attribute__((aligned(16))) bf16 Bt[2][64][32];

    const int tid  = threadIdx.x;
    const int lane = tid & 31;
    const int wave = tid >> 5;
    const int hsel = (lane >> 4) & 1;
    const int l16  = lane & 15;

    v8f acc[4];
#pragma unroll
    for (int nc = 0; nc < 4; ++nc) acc[nc] = vzero();

    const int arow = tid >> 1;         // 0..127
    const int aseg = (tid & 1) * 16;   // 0 or 16
    const int bkk  = tid >> 3;         // 0..31
    const int bn0  = (tid & 7) * 8;    // 0..56

    const bf16* aptr = Ablk + (size_t)arow * lda + aseg;  // + kb
    const unsigned adst[2] = { lds_off(&As[0][arow][aseg]), lds_off(&As[1][arow][aseg]) };

    const int steps = K / 32;

    // prologue: DMA A(step 0) into buffer 0; stage Bt(step 0)
    ASYNC_COPY_32B(adst[0], aptr);
    {
        bf16x8 brow = *(const bf16x8*)(Bm + (size_t)bkk * ldb + bn0);
#pragma unroll
        for (int j = 0; j < 8; ++j) Bt[0][bn0 + j][bkk] = brow[j];
    }
    WAIT_ASYNC(0);
    __syncthreads();

    for (int s = 0; s < steps; ++s) {
        const int cur = s & 1;
        const bool more = (s + 1 < steps);

        bf16x8 brow;
        if (more) {
            // kick off DMA + B row load for step s+1 (into the other buffers)
            ASYNC_COPY_32B(adst[1 - cur], aptr + (size_t)(s + 1) * 32);
            brow = *(const bf16x8*)(Bm + (size_t)((s + 1) * 32 + bkk) * ldb + bn0);
        }

        // ---- compute step s from As[cur] / Bt[cur] ----
        frag16 a;
        const bf16* ap = &As[cur][wave * 16 + l16][hsel * 8];
        a.h[0] = *(const bf16x8*)ap;
        a.h[1] = *(const bf16x8*)(ap + 16);
#pragma unroll
        for (int nc = 0; nc < 4; ++nc) {
            frag16 bf;
            const bf16* bp = &Bt[cur][nc * 16 + l16][hsel * 8];
            bf.h[0] = *(const bf16x8*)bp;
            bf.h[1] = *(const bf16x8*)(bp + 16);
            acc[nc] = wmma_bf16(a.v, bf.v, acc[nc]);
        }

        if (more) {
#pragma unroll
            for (int j = 0; j < 8; ++j) Bt[1 - cur][bn0 + j][bkk] = brow[j];
        }
        WAIT_ASYNC(0);       // own DMA for s+1 landed
        __syncthreads();     // everyone's DMA + Bt stores visible; reads of cur done
    }

    // epilogue: C element (r, lane) -> row = wave*16 + r + 8*hsel, col = nc*16+l16
    const int mrow = wave * 16 + hsel * 8;
#pragma unroll
    for (int nc = 0; nc < 4; ++nc) {
        const int col = nc * 16 + l16;
        const float bv = HAS_BIAS ? bias[col] : 0.0f;
#pragma unroll
        for (int r = 0; r < 8; ++r) {
            float val = acc[nc][r] * alpha + bv;
            Cblk[(size_t)(mrow + r) * ldc + col] = (OutT)val;
        }
    }
}

// ---------------------------------------------------------------------------
// Kernel 1: fused Q/K/V projection. grid = (S/128, B, 3*H)
// q[b,h] = scale * x[b] @ Wq[h]   (scale folded into Q only)
// ---------------------------------------------------------------------------
__global__ __launch_bounds__(256) void qkv_proj_kernel(
    const bf16* __restrict__ xb,
    const bf16* __restrict__ wq, const bf16* __restrict__ wk, const bf16* __restrict__ wv,
    bf16* __restrict__ q, bf16* __restrict__ k, bf16* __restrict__ v)
{
    const int mt  = blockIdx.x;
    const int b   = blockIdx.y;
    const int z   = blockIdx.z;
    const int sel = z >> 4;     // 0=q, 1=k, 2=v
    const int h   = z & 15;

    const bf16* W = (sel == 0) ? wq : (sel == 1) ? wk : wv;
    bf16*       O = (sel == 0) ? q  : (sel == 1) ? k  : v;

    const bf16* Ablk = xb + ((size_t)b * SEQ + (size_t)mt * 128) * EMBED;
    const bf16* Bm   = W + (size_t)h * EMBED * HEADD;
    bf16* Cblk = O + (((size_t)b * NUMH + h) * SEQ + (size_t)mt * 128) * HEADD;

    const float alpha = (sel == 0) ? 0.125f : 1.0f;  // HEAD_SIZE^-0.5 = 1/8
    gemm_block_128x64<bf16, false>(Ablk, EMBED, Bm, HEADD, Cblk, HEADD, EMBED, nullptr, alpha);
}

// ---------------------------------------------------------------------------
// Kernel 2: causal flash attention. grid = (S/128, H, B), 8 waves/block.
// Each wave: 16 query rows; per 32-key block it async-DMAs the contiguous
// K and V sub-blocks (4 KB each) into per-wave LDS, then exploits in-order
// ASYNCcnt completion: wait<=8 -> K landed (compute scores + softmax while
// V is still in flight), wait 0 -> V landed (P*V WMMAs).
// ---------------------------------------------------------------------------
__global__ __launch_bounds__(256) void attn_kernel(
    const bf16* __restrict__ Q, const bf16* __restrict__ K, const bf16* __restrict__ V,
    bf16* __restrict__ Ob)
{
    __shared__ __attribute__((aligned(16))) bf16 Klds[8][32 * HEADD];  // 4 KB / wave
    __shared__ __attribute__((aligned(16))) bf16 Vlds[8][32 * HEADD];  // 4 KB / wave
    __shared__ __attribute__((aligned(16))) bf16 Plds[8][16][32];      // 1 KB / wave

    const int b    = blockIdx.z;
    const int h    = blockIdx.y;
    const int qt   = blockIdx.x;
    const int lane = threadIdx.x & 31;
    const int wave = threadIdx.x >> 5;
    const int hsel = (lane >> 4) & 1;
    const int l16  = lane & 15;
    const int qbase = qt * 128 + wave * 16;

    const size_t head = ((size_t)b * NUMH + h) * SEQ;
    const bf16* Qh = Q + head * HEADD;
    const bf16* Kh = K + head * HEADD;
    const bf16* Vh = V + head * HEADD;

    const unsigned kdst = lds_off(&Klds[wave][lane * 64]);  // 128 B per lane
    const unsigned vdst = lds_off(&Vlds[wave][lane * 64]);

    // preload Q fragments for the two K-dim steps (head-dim 0-31 and 32-63)
    frag16 aq[2];
    {
        const bf16* qrow = Qh + (size_t)(qbase + l16) * HEADD;
#pragma unroll
        for (int c = 0; c < 2; ++c) {
            aq[c].h[0] = *(const bf16x8*)(qrow + 32 * c + hsel * 16);
            aq[c].h[1] = *(const bf16x8*)(qrow + 32 * c + hsel * 16 + 8);
        }
    }

    v8f o[4];
#pragma unroll
    for (int nc = 0; nc < 4; ++nc) o[nc] = vzero();
    float mi[8], li[8];
#pragma unroll
    for (int r = 0; r < 8; ++r) { mi[r] = -3.0e38f; li[r] = 0.0f; }

    const int jblocks = (qbase + 15) / 32 + 1;  // causal: only j0 <= q_end

    for (int jb = 0; jb < jblocks; ++jb) {
        const int j0 = jb * 32;

        // DMA this key block's K then V rows (each lane: one contiguous row)
        ASYNC_COPY_128B(kdst, Kh + (size_t)j0 * HEADD + lane * 64);
        ASYNC_COPY_128B(vdst, Vh + (size_t)j0 * HEADD + lane * 64);
        WAIT_ASYNC(8);  // first 8 (K block) complete; V still in flight

        // ---- scores: S = Q(16x64) * K^T(64x32), two 16-col tiles ----
        v8f s[2];
#pragma unroll
        for (int nt = 0; nt < 2; ++nt) {
            const bf16* krow = &Klds[wave][(nt * 16 + l16) * 64];
            frag16 b0, b1;
            b0.h[0] = *(const bf16x8*)(krow + hsel * 16);
            b0.h[1] = *(const bf16x8*)(krow + hsel * 16 + 8);
            b1.h[0] = *(const bf16x8*)(krow + 32 + hsel * 16);
            b1.h[1] = *(const bf16x8*)(krow + 32 + hsel * 16 + 8);
            v8f z = vzero();
            z = wmma_bf16(aq[0].v, b0.v, z);
            z = wmma_bf16(aq[1].v, b1.v, z);
            s[nt] = z;
        }

        // ---- causal mask ----
#pragma unroll
        for (int nt = 0; nt < 2; ++nt) {
            const int col = j0 + nt * 16 + l16;
#pragma unroll
            for (int r = 0; r < 8; ++r) {
                const int row = qbase + r + hsel * 8;
                s[nt][r] = (col > row) ? -3.0e38f : s[nt][r];
            }
        }

        // ---- online softmax (per row; row's 16 cols live across a half-wave) ----
#pragma unroll
        for (int r = 0; r < 8; ++r) {
            float t = red_max16(fmaxf(s[0][r], s[1][r]));
            float mnew = fmaxf(mi[r], t);
            float al = __expf(mi[r] - mnew);
            mi[r] = mnew;
            float p0 = __expf(s[0][r] - mnew);
            float p1 = __expf(s[1][r] - mnew);
            li[r] = li[r] * al + red_sum16(p0 + p1);
#pragma unroll
            for (int nc = 0; nc < 4; ++nc) o[nc][r] *= al;
            // write P into per-wave LDS tile in plain [row][col] order
            Plds[wave][r + hsel * 8][l16]      = (bf16)p0;
            Plds[wave][r + hsel * 8][16 + l16] = (bf16)p1;
        }

        asm volatile("s_wait_dscnt 0" ::: "memory");  // LDS writes visible to wave

        // ---- reload P as an A fragment (row = l16, K-chunks per ISA layout) ----
        frag16 ap;
        const bf16* pr = &Plds[wave][l16][hsel * 8];
        ap.h[0] = *(const bf16x8*)pr;
        ap.h[1] = *(const bf16x8*)(pr + 16);

        WAIT_ASYNC(0);  // V block landed

        // ---- O += P(16x32) * V(32x64) ----
#pragma unroll
        for (int nc = 0; nc < 4; ++nc) {
            const int col = nc * 16 + l16;
            frag16 bv;
#pragma unroll
            for (int e = 0; e < 16; ++e) {
                const int kr = (e & 7) + hsel * 8 + (e >> 3) * 16;  // key offset within block
                bv.v[e] = Vlds[wave][kr * 64 + col];
            }
            o[nc] = wmma_bf16(ap.v, bv.v, o[nc]);
        }
    }

    // ---- normalize and store into concatenated-head layout [B,S,E] bf16 ----
#pragma unroll
    for (int r = 0; r < 8; ++r) {
        const float inv = 1.0f / li[r];
        const int row = qbase + r + hsel * 8;
        bf16* dst = Ob + ((size_t)b * SEQ + row) * EMBED + h * HEADD;
#pragma unroll
        for (int nc = 0; nc < 4; ++nc)
            dst[nc * 16 + l16] = (bf16)(o[nc][r] * inv);
    }
}

// ---------------------------------------------------------------------------
// Kernel 3: output projection. grid = (B*S/128, E/64)
// out[8192,1024] = O[8192,1024] @ Wp[1024,1024] + bp
// ---------------------------------------------------------------------------
__global__ __launch_bounds__(256) void out_proj_kernel(
    const bf16* __restrict__ Ob, const bf16* __restrict__ wp,
    const float* __restrict__ bp, float* __restrict__ out)
{
    const int mt = blockIdx.x, nt = blockIdx.y;
    gemm_block_128x64<float, true>(
        Ob + (size_t)mt * 128 * EMBED, EMBED,
        wp + (size_t)nt * 64, EMBED,
        out + (size_t)mt * 128 * EMBED + (size_t)nt * 64, EMBED,
        EMBED, bp + (size_t)nt * 64, 1.0f);
}

// ---------------------------------------------------------------------------
// Host launch
// ---------------------------------------------------------------------------
extern "C" void kernel_launch(void* const* d_in, const int* in_sizes, int n_in,
                              void* d_out, int out_size, void* d_ws, size_t ws_size,
                              hipStream_t stream) {
    const float* x  = (const float*)d_in[0];
    const float* Wq = (const float*)d_in[1];
    const float* Wk = (const float*)d_in[2];
    const float* Wv = (const float*)d_in[3];
    const float* Wp = (const float*)d_in[4];
    const float* bp = (const float*)d_in[5];
    float* out = (float*)d_out;

    const size_t nX  = (size_t)BATCH * SEQ * EMBED;   // 8,388,608
    const size_t nW  = (size_t)NUMH * EMBED * HEADD;  // 1,048,576
    const size_t nWp = (size_t)EMBED * EMBED;         // 1,048,576
    const size_t nQ  = (size_t)BATCH * NUMH * SEQ * HEADD;

    char* w = (char*)d_ws;
    bf16* xb  = (bf16*)w; w += nX  * sizeof(bf16);
    bf16* wqb = (bf16*)w; w += nW  * sizeof(bf16);
    bf16* wkb = (bf16*)w; w += nW  * sizeof(bf16);
    bf16* wvb = (bf16*)w; w += nW  * sizeof(bf16);
    bf16* wpb = (bf16*)w; w += nWp * sizeof(bf16);
    bf16* q   = (bf16*)w; w += nQ  * sizeof(bf16);
    bf16* k   = (bf16*)w; w += nQ  * sizeof(bf16);
    bf16* v   = (bf16*)w; w += nQ  * sizeof(bf16);
    bf16* ob  = (bf16*)w; w += nX  * sizeof(bf16);

    cast_f32_bf16_kernel<<<(nX  + 255) / 256, 256, 0, stream>>>(x,  xb,  (int)nX);
    cast_f32_bf16_kernel<<<(nW  + 255) / 256, 256, 0, stream>>>(Wq, wqb, (int)nW);
    cast_f32_bf16_kernel<<<(nW  + 255) / 256, 256, 0, stream>>>(Wk, wkb, (int)nW);
    cast_f32_bf16_kernel<<<(nW  + 255) / 256, 256, 0, stream>>>(Wv, wvb, (int)nW);
    cast_f32_bf16_kernel<<<(nWp + 255) / 256, 256, 0, stream>>>(Wp, wpb, (int)nWp);

    qkv_proj_kernel<<<dim3(SEQ / 128, BATCH, 3 * NUMH), 256, 0, stream>>>(
        xb, wqb, wkb, wvb, q, k, v);

    attn_kernel<<<dim3(SEQ / 128, NUMH, BATCH), 256, 0, stream>>>(q, k, v, ob);

    out_proj_kernel<<<dim3((BATCH * SEQ) / 128, EMBED / 64), 256, 0, stream>>>(
        ob, wpb, bp, out);
}
